// ImagePatchAttention_63754494542004
// MI455X (gfx1250) — matrix operand order repaired and checked by reading, not compiled, not run on hardware
//
#include <hip/hip_runtime.h>
#include <hip/hip_bf16.h>
#include <math.h>

// Problem constants (from reference): B=16, H=768, N=2048
constexpr int Bb = 16;
constexpr int Hh = 768;
constexpr int Nn = 2048;

constexpr int KT    = 32;            // keys per streamed tile
constexpr int QROWS = 16;            // query rows per workgroup
constexpr int NWAVE = 8;             // waves per workgroup (wave32)
constexpr int HPW   = Hh / NWAVE;    // 96 h-dims owned per wave
constexpr int KROW  = Hh + 8;        // padded halves per LDS row (conflict-free frag reads)
constexpr int PROW  = KT + 8;        // padded halves per P row
constexpr int NT    = Nn / KT;       // 64 key tiles

typedef __attribute__((ext_vector_type(16))) _Float16 v16h;
typedef __attribute__((ext_vector_type(8)))  float    v8f;

union Frag16 { v16h h; uint4 u[2]; };

__global__ __launch_bounds__(256)
void ipa_flash_wmma(const float* __restrict__ q,
                    const float* __restrict__ k,
                    const float* __restrict__ v,
                    const float* __restrict__ pe,
                    float* __restrict__ out)
{
    // ---- LDS (~141 KB total; 2 workgroups still fit a 320 KB WGP) ----
    __shared__ __align__(16) _Float16 Qs[QROWS * KROW];          // 24,832 B  [n][h]
    __shared__ __align__(16) _Float16 Ks[2][KT * KROW];          // 99,328 B  [m][h] x2 (double buffer)
    __shared__ __align__(16) float    Spart[NWAVE * 2 * 8 * 32]; // 16,384 B  [wave][t][r][lane]
    __shared__ __align__(16) float    Sfin[2 * 8 * 32];          //  2,048 B  [t][r][lane]
    __shared__ __align__(16) _Float16 Ps[QROWS * PROW];          //  1,280 B  [n][m]

    const int tid  = threadIdx.x;
    const int wave = tid >> 5;
    const int lane = tid & 31;
    const int hi   = lane >> 4;      // half-wave (0/1)
    const int lj   = lane & 15;

    const int b  = blockIdx.y;
    const int n0 = blockIdx.x * QROWS;

    const size_t base = (size_t)b * Hh * Nn;
    const float scale = 0.036084391824351615f;   // 1/sqrt(768)

    // ---- stage K' = key+key_pe (f16, transposed [m][h]) into a buffer ----
    auto stageK = [&](int m0, _Float16* dst) {
        // 2h x 4m blocks: float4 global loads, packed b32 LDS stores
        for (int p = tid; p < (Hh / 2) * (KT / 4); p += 256) {   // 3072 items
            const int m  = (p & 7) * 4;
            const int h  = (p >> 3) * 2;
            const size_t g0 = base + (size_t)h * Nn + m0 + m;
            const float4 k0 = *(const float4*)(k  + g0);
            const float4 k1 = *(const float4*)(k  + g0 + Nn);
            const float4 e0 = *(const float4*)(pe + g0);
            const float4 e1 = *(const float4*)(pe + g0 + Nn);
            const float a0[4] = {k0.x + e0.x, k0.y + e0.y, k0.z + e0.z, k0.w + e0.w};
            const float a1[4] = {k1.x + e1.x, k1.y + e1.y, k1.z + e1.z, k1.w + e1.w};
            #pragma unroll
            for (int j = 0; j < 4; ++j) {
                _Float16* d = &dst[(m + j) * KROW + h];
                d[0] = (_Float16)a0[j];
                d[1] = (_Float16)a1[j];
            }
        }
    };

    // ---- stage Q block once, transposed to [n][h] halves ----
    for (int p = tid; p < (Hh / 2) * (QROWS / 4); p += 256) {    // 1536 items
        const int n  = (p & 3) * 4;
        const int h  = (p >> 2) * 2;
        const size_t g0 = base + (size_t)h * Nn + n0 + n;
        const float4 q0 = *(const float4*)(q + g0);
        const float4 q1 = *(const float4*)(q + g0 + Nn);
        const float a0[4] = {q0.x, q0.y, q0.z, q0.w};
        const float a1[4] = {q1.x, q1.y, q1.z, q1.w};
        #pragma unroll
        for (int j = 0; j < 4; ++j) {
            _Float16* d = &Qs[(n + j) * KROW + h];
            d[0] = (_Float16)a0[j];
            d[1] = (_Float16)a1[j];
        }
    }

    // ---- per-wave O accumulator lives in registers: 6 tiles x v8f = 48 VGPRs ----
    v8f Oacc[6];
    #pragma unroll
    for (int t = 0; t < 6; ++t)
        #pragma unroll
        for (int r = 0; r < 8; ++r) Oacc[t][r] = 0.0f;

    float m_run[8], l_run[8];
    #pragma unroll
    for (int r = 0; r < 8; ++r) { m_run[r] = -3.0e38f; l_run[r] = 0.0f; }

    stageK(0, Ks[0]);
    __syncthreads();

    // ================= stream over key tiles =================
    for (int kt = 0; kt < NT; ++kt) {
        const int m0 = kt * KT;
        _Float16* KsCur = Ks[kt & 1];
        _Float16* KsNxt = Ks[(kt + 1) & 1];

        // ---- stage next K' tile into the alternate buffer (overlaps compute) ----
        if (kt + 1 < NT) stageK(m0 + KT, KsNxt);

        // ---- prefetch tile kt+2 into L2 (global_prefetch_b8), full row coverage ----
        if (kt + 2 < NT) {
            #pragma unroll
            for (int r3 = 0; r3 < 3; ++r3) {
                const size_t gn = base + (size_t)(tid + r3 * 256) * Nn + m0 + 2 * KT;
                __builtin_prefetch(k  + gn, 0, 1);
                __builtin_prefetch(pe + gn, 0, 1);
            }
        }

        // ---- partial S over this wave's 96-dim h slice ----
        v8f accS[2];
        #pragma unroll
        for (int r = 0; r < 8; ++r) { accS[0][r] = 0.0f; accS[1][r] = 0.0f; }

        #pragma unroll
        for (int c = 0; c < 3; ++c) {
            const int hstart = wave * HPW + c * 32;
            Frag16 afr;
            {
                const char* ap = (const char*)Qs + lj * (KROW * 2) + hstart * 2 + (hi << 4);
                afr.u[0] = *(const uint4*)(ap);
                afr.u[1] = *(const uint4*)(ap + 32);
            }
            #pragma unroll
            for (int t = 0; t < 2; ++t) {
                const int m = t * 16 + lj;
                Frag16 bfr;
                const char* bp = (const char*)KsCur + m * (KROW * 2) + hstart * 2 + (hi << 4);
                bfr.u[0] = *(const uint4*)(bp);
                bfr.u[1] = *(const uint4*)(bp + 32);
                accS[t] = __builtin_amdgcn_wmma_f32_16x16x32_f16(
                    false, afr.h, false, bfr.h, (short)0, accS[t], false, false);
            }
        }
        #pragma unroll
        for (int t = 0; t < 2; ++t)
            #pragma unroll
            for (int r = 0; r < 8; ++r)
                Spart[((wave * 2 + t) * 8 + r) * 32 + lane] = accS[t][r];
        __syncthreads();   // barrier 1: partials ready (also: KsNxt fully staged)

        // ---- reduce 8 wave-partials -> Sfin (all 256 threads) ----
        for (int e = tid; e < 512; e += 256) {
            float s = Spart[e];
            #pragma unroll
            for (int w2 = 1; w2 < 8; ++w2) s += Spart[w2 * 512 + e];
            Sfin[e] = s;
        }
        __syncthreads();   // barrier 2: Sfin ready

        // ---- online softmax (redundant & deterministic in every wave) ----
        float pbuf[2][8];
        float alpha[8];
        #pragma unroll
        for (int r = 0; r < 8; ++r) {
            const float s0 = Sfin[r * 32 + lane] * scale;
            const float s1 = Sfin[256 + r * 32 + lane] * scale;
            float mx = fmaxf(s0, s1);
            #pragma unroll
            for (int off = 1; off < 16; off <<= 1)
                mx = fmaxf(mx, __shfl_xor(mx, off, 32));
            const float mnew = fmaxf(m_run[r], mx);
            alpha[r] = __expf(m_run[r] - mnew);
            m_run[r] = mnew;
            const float p0 = __expf(s0 - mnew);
            const float p1 = __expf(s1 - mnew);
            pbuf[0][r] = p0; pbuf[1][r] = p1;
            float ps = p0 + p1;
            #pragma unroll
            for (int off = 1; off < 16; off <<= 1)
                ps += __shfl_xor(ps, off, 32);
            l_run[r] = l_run[r] * alpha[r] + ps;
        }

        // ---- rescale O accumulator (registers only) ----
        #pragma unroll
        for (int t = 0; t < 6; ++t)
            #pragma unroll
            for (int r = 0; r < 8; ++r)
                Oacc[t][r] *= alpha[r];

        // ---- wave 0 publishes P tile in A-fragment [n][m] layout ----
        if (wave == 0) {
            #pragma unroll
            for (int t = 0; t < 2; ++t)
                #pragma unroll
                for (int r = 0; r < 8; ++r) {
                    const int n = r + (hi << 3);
                    const int m = t * 16 + lj;
                    Ps[n * PROW + m] = (_Float16)pbuf[t][r];
                }
        }
        __syncthreads();   // barrier 3: Ps ready

        // ---- O += P x V^T for this wave's 6 h-tiles (K = 32 keys) ----
        Frag16 afr;
        {
            const char* ap = (const char*)Ps + lj * (PROW * 2) + (hi << 5);
            afr.u[0] = *(const uint4*)(ap);
            afr.u[1] = *(const uint4*)(ap + 16);
        }
        #pragma unroll
        for (int t = 0; t < 6; ++t) {
            const int h = wave * HPW + t * 16 + lj;
            const float* vp = v + base + (size_t)h * Nn + m0 + (hi << 4);
            Frag16 bfr;
            #pragma unroll
            for (int j = 0; j < 4; ++j) {
                const float4 f = *(const float4*)(vp + j * 4);
                bfr.h[j * 4 + 0] = (_Float16)f.x;
                bfr.h[j * 4 + 1] = (_Float16)f.y;
                bfr.h[j * 4 + 2] = (_Float16)f.z;
                bfr.h[j * 4 + 3] = (_Float16)f.w;
            }
            Oacc[t] = __builtin_amdgcn_wmma_f32_16x16x32_f16(
                false, afr.h, false, bfr.h, (short)0, Oacc[t], false, false);
        }
        // no trailing barrier: next-iter staging writes KsCur, whose last readers
        // (this iteration's S phase) completed before barrier 1 of this iteration.
    }

    // ---- normalize + write out: out[b][h][n] = O[n][h] / l[n] ----
    float inv_l[8];
    #pragma unroll
    for (int r = 0; r < 8; ++r) inv_l[r] = 1.0f / l_run[r];

    #pragma unroll
    for (int t = 0; t < 6; ++t) {
        const int h = wave * HPW + t * 16 + lj;
        float vals[8];
        #pragma unroll
        for (int r = 0; r < 8; ++r) vals[r] = Oacc[t][r] * inv_l[r];
        float* op = out + base + (size_t)h * Nn + n0 + (hi << 3);
        *(float4*)(op)     = make_float4(vals[0], vals[1], vals[2], vals[3]);
        *(float4*)(op + 4) = make_float4(vals[4], vals[5], vals[6], vals[7]);
    }
}

extern "C" void kernel_launch(void* const* d_in, const int* in_sizes, int n_in,
                              void* d_out, int out_size, void* d_ws, size_t ws_size,
                              hipStream_t stream) {
    (void)in_sizes; (void)n_in; (void)d_ws; (void)ws_size; (void)out_size;
    const float* q  = (const float*)d_in[0];   // query
    const float* k  = (const float*)d_in[1];   // key
    const float* v  = (const float*)d_in[2];   // value
    const float* pe = (const float*)d_in[3];   // key_pe
    float* out = (float*)d_out;

    dim3 grid(Nn / QROWS, Bb);   // 128 x 16 blocks; batch slow-varying for L2 reuse
    dim3 block(256);             // 8 wave32 waves
    ipa_flash_wmma<<<grid, block, 0, stream>>>(q, k, v, pe, out);
}
